// TemporalMemoryModule_21492016349926
// MI455X (gfx1250) — matrix operand order repaired and checked by reading, not compiled, not run on hardware
//
#include <hip/hip_runtime.h>
#include <hip/hip_bf16.h>

typedef __attribute__((ext_vector_type(16))) __bf16 v16bf;
typedef __attribute__((ext_vector_type(8)))  float  v8f;

#define MEMD 128
#define MSGD 128
#define K1   416   // 400 padded to 13*32
#define K1T  13
#define K2T  4     // 128/32

union AFrag { v16bf v; unsigned int u[8]; };

static __device__ __forceinline__ unsigned short f2bf(float f) {
  unsigned int u = __float_as_uint(f);
  unsigned int r = u + 0x7FFFu + ((u >> 16) & 1u);   // RNE
  return (unsigned short)(r >> 16);
}

static __device__ __forceinline__ float sigmoidf_(float x) {
  return 1.0f / (1.0f + expf(-x));
}

// A-matrix 16x32 bf16 fragment from LDS (row-major ushort, stride lda elems).
// lane<16: M=lane,   VGPR0-3: K=0..7,  VGPR4-7: K=16..23
// lane>=16:M=lane-16,VGPR0-3: K=8..15, VGPR4-7: K=24..31
static __device__ __forceinline__ v16bf load_a_frag(const unsigned short* A, int lda,
                                                    int row0, int k0, int lane) {
  int m = row0 + (lane & 15);
  int khalf = (lane >> 4) << 3;
  const unsigned short* p = A + m * lda + k0 + khalf;
  AFrag f;
#pragma unroll
  for (int v = 0; v < 4; ++v) f.u[v] = *(const unsigned int*)(p + 2 * v);
#pragma unroll
  for (int v = 0; v < 4; ++v) f.u[4 + v] = *(const unsigned int*)(p + 16 + 2 * v);
  return f.v;
}

// B-matrix 32x16 fragment from packed global weights: each lane owns 16
// contiguous bf16 (32B). Packed index: (((nt*KT)+kt)*32+lane)*16 + j
static __device__ __forceinline__ v16bf load_b_frag(const unsigned short* P, int kt, int nt,
                                                    int KT, int lane) {
  const unsigned short* p = P + ((((size_t)nt * KT) + kt) * 32 + lane) * 16;
  return *(const v16bf*)p;
}

static __device__ __forceinline__ v8f wmma_bf16(v16bf a, v16bf b, v8f c) {
  return __builtin_amdgcn_wmma_f32_16x16x32_bf16(false, a, false, b, (short)0, c, false, false);
}

// ---------------- setup kernels ----------------

__global__ void tgn_init_kernel(int* last_pos, int* count, int N) {
  int i = blockIdx.x * blockDim.x + threadIdx.x;
  if (i < N) last_pos[i] = -1;
  if (i == 0) *count = 0;
}

__global__ void tgn_scatter_max_kernel(const int* __restrict__ src, const int* __restrict__ dst,
                                       int* last_pos, int E) {
  int i = blockIdx.x * blockDim.x + threadIdx.x;
  if (i < 2 * E) {
    int node = (i < E) ? src[i] : dst[i - E];
    atomicMax(&last_pos[node], i);
  }
}

__global__ void tgn_compact_kernel(const int* __restrict__ last_pos, int* nodes_list,
                                   int* count, int N) {
  int i = blockIdx.x * blockDim.x + threadIdx.x;
  if (i < N && last_pos[i] >= 0) {
    int p = atomicAdd(count, 1);
    nodes_list[p] = i;
  }
}

// Pack B [K x N] (bf16) into per-lane-contiguous WMMA fragments.
// transpose==0: B[k][n] = src[k*ld + n]; transpose==1: B[k][n] = src[n*ld + k]
__global__ void tgn_pack_b_kernel(const float* __restrict__ src, unsigned short* __restrict__ dst,
                                  int K, int N, int srcK, int transpose, int ld) {
  int KT = K >> 5, NT = N >> 4;
  int total = KT * NT * 512;
  for (int idx = blockIdx.x * blockDim.x + threadIdx.x; idx < total;
       idx += gridDim.x * blockDim.x) {
    int j    = idx & 15;
    int lane = (idx >> 4) & 31;
    int tile = idx >> 9;           // c*KT + t
    int t = tile % KT;
    int c = tile / KT;
    int n = c * 16 + (lane & 15);
    int k = t * 32 + ((lane >> 4) << 3) + (j & 7) + ((j >> 3) << 4);
    float val = 0.0f;
    if (k < srcK)
      val = transpose ? src[(size_t)n * ld + k] : src[(size_t)k * ld + n];
    dst[idx] = f2bf(val);
  }
}

// ---------------- message MLP for winner rows ----------------
// Block: 64 rows x 128 cols. 8 waves: wave = (wr in 0..3 row strip) x (wc in 0..1 col half).
__launch_bounds__(256)
__global__ void tgn_msg_kernel(const int* __restrict__ src_ids, const int* __restrict__ dst_ids,
                               const float* __restrict__ edge_feat,
                               const float* __restrict__ timestamps,
                               const float* __restrict__ memory, const float* __restrict__ lut,
                               const float* __restrict__ time_w, const float* __restrict__ time_phi,
                               const float* __restrict__ b1, const float* __restrict__ b2,
                               const unsigned short* __restrict__ pW1,
                               const unsigned short* __restrict__ pW2,
                               const int* __restrict__ nodes_list,
                               const int* __restrict__ last_pos,
                               const int* __restrict__ count,
                               unsigned short* __restrict__ agg_out, int E) {
  __shared__ unsigned short sm[64 * K1];   // 53248 B; reused as h-tile after GEMM1
  int cnt = *count;
  int rowBase = blockIdx.x * 64;
  if (rowBase >= cnt) return;

  int tid = threadIdx.x;
  // ---- build A tile: [self_mem | other_mem | edge_feat | time2vec | pad0] ----
  {
    int r = tid >> 2;          // 0..63
    int sub = tid & 3;
    int g = rowBase + r;
    bool valid = g < cnt;
    int e = 0, self = 0, other = 0;
    float dt = 0.0f;
    if (valid) {
      int node = nodes_list[g];
      int pos = last_pos[node];
      bool isSrc = pos < E;
      e = isSrc ? pos : pos - E;
      int s = src_ids[e], d = dst_ids[e];
      self  = isSrc ? s : d;
      other = isSrc ? d : s;
      dt = timestamps[e] - lut[self];
    }
    unsigned short* row = sm + r * K1;
    for (int c = sub; c < K1; c += 4) {
      float v = 0.0f;
      if (valid) {
        if (c < 128)      v = memory[(size_t)self * MEMD + c];
        else if (c < 256) v = memory[(size_t)other * MEMD + (c - 128)];
        else if (c < 384) v = edge_feat[(size_t)e * MSGD + (c - 256)];
        else if (c < 400) {
          int j = c - 384;
          float wt = dt * time_w[j] + time_phi[j];
          v = (j == 0) ? wt : sinf(wt);
        }
      }
      row[c] = f2bf(v);
    }
  }
  __syncthreads();

  int w = tid >> 5, lane = tid & 31;
  int wr = w & 3;       // row strip
  int wc = w >> 2;      // col half

  // ---- GEMM1: h = relu(A @ W1 + b1), K = 416 ----
  v8f acc[4] = {};
  for (int t = 0; t < K1T; ++t) {
    v16bf a = load_a_frag(sm, K1, wr * 16, t * 32, lane);
#pragma unroll
    for (int ct = 0; ct < 4; ++ct) {
      v16bf b = load_b_frag(pW1, t, wc * 4 + ct, K1T, lane);
      acc[ct] = wmma_bf16(a, b, acc[ct]);
    }
  }
  __syncthreads();                 // everyone done reading sm

  unsigned short* hs = sm;         // alias: h tile [64][136] bf16
  const int LDH = 136;
  {
    int mrow = wr * 16 + 8 * (lane >> 4);
#pragma unroll
    for (int ct = 0; ct < 4; ++ct) {
      int n = wc * 64 + ct * 16 + (lane & 15);
      float bb = b1[n];
#pragma unroll
      for (int i = 0; i < 8; ++i) {
        float v = acc[ct][i] + bb;
        hs[(mrow + i) * LDH + n] = f2bf(v > 0.0f ? v : 0.0f);
      }
    }
  }
  __syncthreads();

  // ---- GEMM2: msg = h @ W2 + b2, K = 128 ----
  v8f acc2[4] = {};
  for (int t = 0; t < K2T; ++t) {
    v16bf a = load_a_frag(hs, LDH, wr * 16, t * 32, lane);
#pragma unroll
    for (int ct = 0; ct < 4; ++ct) {
      v16bf b = load_b_frag(pW2, t, wc * 4 + ct, K2T, lane);
      acc2[ct] = wmma_bf16(a, b, acc2[ct]);
    }
  }
  {
    int mbase = wr * 16 + 8 * (lane >> 4);
#pragma unroll
    for (int ct = 0; ct < 4; ++ct) {
      int n = wc * 64 + ct * 16 + (lane & 15);
      float bb = b2[n];
#pragma unroll
      for (int i = 0; i < 8; ++i) {
        int g = rowBase + mbase + i;
        if (g < cnt) agg_out[(size_t)g * MSGD + n] = f2bf(acc2[ct][i] + bb);
      }
    }
  }
}

// ---------------- GRU over winner rows ----------------
// Block: 32 rows. 16 wave-tasks = (2 row strips) x (8 col tiles); each wave does 2.
// Task = 3 gi tiles (r,z,n gate cols) + 3 gh tiles, fused gate epilogue.
__launch_bounds__(256)
__global__ void tgn_gru_kernel(const float* __restrict__ memory,
                               const unsigned short* __restrict__ agg,
                               const unsigned short* __restrict__ pWih,
                               const unsigned short* __restrict__ pWhh,
                               const float* __restrict__ b_ih, const float* __restrict__ b_hh,
                               const int* __restrict__ nodes_list,
                               const int* __restrict__ count,
                               float* __restrict__ mem_out) {
  __shared__ unsigned short sa[32 * MEMD];   // agg tile (bf16)
  __shared__ unsigned short sh[32 * MEMD];   // memory tile (bf16)
  int cnt = *count;
  int rowBase = blockIdx.x * 32;
  if (rowBase >= cnt) return;

  int tid = threadIdx.x;
  for (int idx = tid; idx < 32 * MEMD; idx += 256) {
    int r = idx >> 7, c = idx & 127;
    int g = rowBase + r;
    unsigned short va = 0, vh = 0;
    if (g < cnt) {
      va = agg[(size_t)g * MSGD + c];
      int node = nodes_list[g];
      vh = f2bf(memory[(size_t)node * MEMD + c]);
    }
    sa[idx] = va;
    sh[idx] = vh;
  }
  __syncthreads();

  int w = tid >> 5, lane = tid & 31;
  for (int task = w; task < 16; task += 8) {
    int s  = task >> 3;     // row strip 0..1
    int ct = task & 7;      // output col tile 0..7
    v8f accI[3] = {};
    v8f accH[3] = {};
    for (int t = 0; t < K2T; ++t) {
      v16bf aA = load_a_frag(sa, MEMD, s * 16, t * 32, lane);
      v16bf aH = load_a_frag(sh, MEMD, s * 16, t * 32, lane);
#pragma unroll
      for (int gg = 0; gg < 3; ++gg) {
        v16bf bI = load_b_frag(pWih, t, ct + 8 * gg, K2T, lane);
        accI[gg] = wmma_bf16(aA, bI, accI[gg]);
        v16bf bH = load_b_frag(pWhh, t, ct + 8 * gg, K2T, lane);
        accH[gg] = wmma_bf16(aH, bH, accH[gg]);
      }
    }
    int col = ct * 16 + (lane & 15);
    float bi_r = b_ih[col], bi_z = b_ih[col + 128], bi_n = b_ih[col + 256];
    float bh_r = b_hh[col], bh_z = b_hh[col + 128], bh_n = b_hh[col + 256];
    int mbase = s * 16 + 8 * (lane >> 4);
#pragma unroll
    for (int i = 0; i < 8; ++i) {
      int g = rowBase + mbase + i;
      if (g < cnt) {
        int node = nodes_list[g];
        float hprev = memory[(size_t)node * MEMD + col];
        float r  = sigmoidf_((accI[0][i] + bi_r) + (accH[0][i] + bh_r));
        float z  = sigmoidf_((accI[1][i] + bi_z) + (accH[1][i] + bh_z));
        float nn = tanhf((accI[2][i] + bi_n) + r * (accH[2][i] + bh_n));
        mem_out[(size_t)node * MEMD + col] = (1.0f - z) * nn + z * hprev;
      }
    }
  }
}

// ---------------- launch ----------------

extern "C" void kernel_launch(void* const* d_in, const int* in_sizes, int n_in,
                              void* d_out, int out_size, void* d_ws, size_t ws_size,
                              hipStream_t stream) {
  const int*   src_ids    = (const int*)d_in[0];
  const int*   dst_ids    = (const int*)d_in[1];
  const float* edge_feat  = (const float*)d_in[2];
  const float* timestamps = (const float*)d_in[3];
  const float* memory     = (const float*)d_in[4];
  const float* lut        = (const float*)d_in[5];
  const float* time_w     = (const float*)d_in[6];
  const float* time_phi   = (const float*)d_in[7];
  const float* W1         = (const float*)d_in[8];
  const float* b1         = (const float*)d_in[9];
  const float* W2         = (const float*)d_in[10];
  const float* b2         = (const float*)d_in[11];
  const float* W_ih       = (const float*)d_in[12];
  const float* W_hh       = (const float*)d_in[13];
  const float* b_ih       = (const float*)d_in[14];
  const float* b_hh       = (const float*)d_in[15];
  (void)n_in; (void)out_size; (void)ws_size;

  const int E = in_sizes[0];
  const int N = in_sizes[4] / MEMD;

  // workspace layout (256B aligned slices)
  char* ws = (char*)d_ws;
  size_t off = 0;
  auto take = [&](size_t bytes) {
    char* p = ws + off;
    off = (off + bytes + 255) & ~(size_t)255;
    return p;
  };
  int* d_last            = (int*)take((size_t)N * 4);
  int* d_list            = (int*)take((size_t)N * 4);
  int* d_cnt             = (int*)take(4);
  unsigned short* pW1    = (unsigned short*)take((size_t)K1T * 8 * 512 * 2);
  unsigned short* pW2    = (unsigned short*)take((size_t)K2T * 8 * 512 * 2);
  unsigned short* pWih   = (unsigned short*)take((size_t)K2T * 24 * 512 * 2);
  unsigned short* pWhh   = (unsigned short*)take((size_t)K2T * 24 * 512 * 2);
  unsigned short* d_agg  = (unsigned short*)take((size_t)N * MSGD * 2);

  tgn_init_kernel<<<(N + 255) / 256, 256, 0, stream>>>(d_last, d_cnt, N);

  tgn_pack_b_kernel<<<208, 256, 0, stream>>>(W1,  pW1,  K1,  128, 400, 0, 128);
  tgn_pack_b_kernel<<< 64, 256, 0, stream>>>(W2,  pW2,  128, 128, 128, 0, 128);
  tgn_pack_b_kernel<<<192, 256, 0, stream>>>(W_ih, pWih, 128, 384, 128, 1, 128);
  tgn_pack_b_kernel<<<192, 256, 0, stream>>>(W_hh, pWhh, 128, 384, 128, 1, 128);

  tgn_scatter_max_kernel<<<(2 * E + 255) / 256, 256, 0, stream>>>(src_ids, dst_ids, d_last, E);
  tgn_compact_kernel<<<(N + 255) / 256, 256, 0, stream>>>(d_last, d_list, d_cnt, N);

  // default output = old memory (non-present rows keep it)
  hipMemcpyAsync(d_out, memory, (size_t)N * MEMD * sizeof(float),
                 hipMemcpyDeviceToDevice, stream);

  tgn_msg_kernel<<<(N + 63) / 64, 256, 0, stream>>>(
      src_ids, dst_ids, edge_feat, timestamps, memory, lut, time_w, time_phi,
      b1, b2, pW1, pW2, d_list, d_last, d_cnt, d_agg, E);

  tgn_gru_kernel<<<(N + 31) / 32, 256, 0, stream>>>(
      memory, d_agg, pWih, pWhh, b_ih, b_hh, d_list, d_cnt, (float*)d_out);
}
